// MultiHeadAttention_15582141350299
// MI455X (gfx1250) — compile-verified, matrix operand order checked by reading, not run on hardware
//
#include <hip/hip_runtime.h>
#include <hip/hip_bf16.h>

typedef __attribute__((ext_vector_type(16))) __bf16 v16bf;
typedef __attribute__((ext_vector_type(8)))  float  v8f;

#define NFEAT 1024
#define NHEAD 16
#define DK    64
#define BB    2
#define SS    2048

#if defined(__AMDGCN__) && __has_builtin(__builtin_amdgcn_global_load_async_to_lds_b128) && __has_builtin(__builtin_amdgcn_s_wait_asynccnt)
#define USE_ASYNC_LDS 1
typedef int v4i __attribute__((vector_size(16)));            // matches builtin param
typedef __attribute__((address_space(1))) v4i g_v4i;
typedef __attribute__((address_space(3))) v4i l_v4i;
#else
#define USE_ASYNC_LDS 0
#endif

__device__ __forceinline__ unsigned short f32_to_bf16(float f) {
  unsigned int u = __float_as_uint(f);
  u += 0x7fffu + ((u >> 16) & 1u);   // round-to-nearest-even
  return (unsigned short)(u >> 16);
}

union ABu { v16bf v; uint4 q[2]; unsigned int u[8]; };

// Load a 16x32 bf16 A-tile (or 32x16 B-tile given B^T row-major storage).
// Per-lane layout (CDNA5 ISA 7.12.2, 16-bit A 16x32):
//   lane l: row r = l&15; K-chunks of 8 at K = (l>=16?8:0) and 16+(l>=16?8:0)
__device__ __forceinline__ v16bf load_tile(const unsigned short* p, int ld) {
  int l  = threadIdx.x & 31;
  int r  = l & 15;
  int c0 = (l >> 4) * 8;
  ABu u;
  u.q[0] = *(const uint4*)(p + (size_t)r * ld + c0);
  u.q[1] = *(const uint4*)(p + (size_t)r * ld + c0 + 16);
  return u.v;
}

__device__ __forceinline__ v8f wmma_bf16(v16bf a, v16bf b, v8f c) {
  return __builtin_amdgcn_wmma_f32_16x16x32_bf16(false, a, false, b, (short)0, c,
                                                 false, false);
}

__global__ void cvt_f32_bf16(const float* __restrict__ in,
                             unsigned short* __restrict__ out, int n) {
  int i = blockIdx.x * blockDim.x + threadIdx.x;
  int stride = gridDim.x * blockDim.x;
  for (; i < n; i += stride) out[i] = f32_to_bf16(in[i]);
}

// Y[m,n] = sum_k A[m,k] * W[n,k] + bias[n]
// Block = 4 waves: 64 rows x 64 cols. Weight chunk (64n x 32k bf16 = 4KB)
// staged in LDS, double-buffered via global_load_async_to_lds_b128.
// mode 0: Q -> (B,H,S,D) bf16, scaled by 0.125 (1/sqrt(DK))
// mode 1: K -> (B,H,S,D) bf16
// mode 2: V -> (B,H,D,S) bf16 (transposed so attention B-operand is K-contiguous)
// mode 3: fp32 row-major (final output projection)
__global__ void proj_wmma(const unsigned short* __restrict__ A,
                          const unsigned short* __restrict__ W,
                          const float* __restrict__ bias,
                          unsigned short* __restrict__ out_bf16,
                          float* __restrict__ out_f32,
                          int mode) {
  __shared__ alignas(16) unsigned short Bbuf[2][64][32];
  const int NT = NFEAT / 64;               // 16 column blocks
  int tid   = threadIdx.x;
  int w     = tid >> 5;
  int l     = tid & 31;
  int mbase = (blockIdx.x / NT) * 64 + w * 16;
  int nbase = (blockIdx.x % NT) * 64;

  // copy mapping: 128 threads cover 64 rows x 64B (4KB) per chunk, 2 x b128 each
  int crow = tid >> 2;                     // 0..31
  int cseg = (tid & 3) * 8;                // element offset in k-chunk

  const unsigned short* Abase = A + (size_t)mbase * NFEAT;
  const unsigned short* Wbase = W + (size_t)nbase * NFEAT;

  v8f acc[4] = {};
  const int NITER = NFEAT / 32;            // 32 k-chunks

#if USE_ASYNC_LDS
  // prologue: kick chunk 0 into buffer 0
  {
    const unsigned short* g0 = Wbase + (size_t)crow * NFEAT + cseg;
    const unsigned short* g1 = Wbase + (size_t)(crow + 32) * NFEAT + cseg;
    __builtin_amdgcn_global_load_async_to_lds_b128(
        (g_v4i*)g0, (l_v4i*)&Bbuf[0][crow][cseg], 0, 0);
    __builtin_amdgcn_global_load_async_to_lds_b128(
        (g_v4i*)g1, (l_v4i*)&Bbuf[0][crow + 32][cseg], 0, 0);
  }
  for (int i = 0; i < NITER; ++i) {
    int cur = i & 1;
    if (i + 1 < NITER) {
      int kc = (i + 1) * 32;
      const unsigned short* g0 = Wbase + (size_t)crow * NFEAT + kc + cseg;
      const unsigned short* g1 = Wbase + (size_t)(crow + 32) * NFEAT + kc + cseg;
      __builtin_amdgcn_global_load_async_to_lds_b128(
          (g_v4i*)g0, (l_v4i*)&Bbuf[1 - cur][crow][cseg], 0, 0);
      __builtin_amdgcn_global_load_async_to_lds_b128(
          (g_v4i*)g1, (l_v4i*)&Bbuf[1 - cur][crow + 32][cseg], 0, 0);
      __builtin_amdgcn_s_wait_asynccnt(2);   // async loads retire in order
    } else {
      __builtin_amdgcn_s_wait_asynccnt(0);
    }
    __syncthreads();
    v16bf a = load_tile(Abase + i * 32, NFEAT);
#pragma unroll
    for (int t = 0; t < 4; ++t) {
      ABu u;
      const unsigned short* bp = &Bbuf[cur][t * 16 + (l & 15)][(l >> 4) * 8];
      u.q[0] = *(const uint4*)bp;
      u.q[1] = *(const uint4*)(bp + 16);
      acc[t] = wmma_bf16(a, u.v, acc[t]);
    }
    __syncthreads();   // all waves done with Bbuf[cur] before it is refilled
  }
#else
  for (int i = 0; i < NITER; ++i) {
    int kc = i * 32;
    uint4 r0 = *(const uint4*)(Wbase + (size_t)crow * NFEAT + kc + cseg);
    uint4 r1 = *(const uint4*)(Wbase + (size_t)(crow + 32) * NFEAT + kc + cseg);
    *(uint4*)&Bbuf[0][crow][cseg]      = r0;
    *(uint4*)&Bbuf[0][crow + 32][cseg] = r1;
    __syncthreads();
    v16bf a = load_tile(Abase + kc, NFEAT);
#pragma unroll
    for (int t = 0; t < 4; ++t) {
      ABu u;
      const unsigned short* bp = &Bbuf[0][t * 16 + (l & 15)][(l >> 4) * 8];
      u.q[0] = *(const uint4*)bp;
      u.q[1] = *(const uint4*)(bp + 16);
      acc[t] = wmma_bf16(a, u.v, acc[t]);
    }
    __syncthreads();
  }
#endif

  int col   = l & 15;
  int rbase = (l >> 4) * 8;
#pragma unroll
  for (int t = 0; t < 4; ++t) {
    int n = nbase + t * 16 + col;
    float bv = bias[n];
#pragma unroll
    for (int j = 0; j < 8; ++j) {
      int m = mbase + rbase + j;
      float y = acc[t][j] + bv;
      if (mode <= 2) {
        int b = m >> 11, s = m & (SS - 1);
        int h = n >> 6,  d = n & 63;
        if (mode == 0) y *= 0.125f;
        size_t idx = (mode == 2)
          ? ((size_t)(b * NHEAD + h) * DK + d) * SS + s
          : ((size_t)(b * NHEAD + h) * SS + s) * DK + d;
        out_bf16[idx] = f32_to_bf16(y);
      } else {
        out_f32[(size_t)m * NFEAT + n] = y;
      }
    }
  }
}

// Flash-attention: one wave handles (b,h, 16 query rows); k-tiles of 32.
__global__ void attn_fwd(const unsigned short* __restrict__ Qh,
                         const unsigned short* __restrict__ Kh,
                         const unsigned short* __restrict__ VhT,
                         const unsigned char* __restrict__ mask,
                         unsigned short* __restrict__ X) {
  __shared__ unsigned short Plds[4][32][16];   // per-wave, column-major P tile
  int w  = threadIdx.x >> 5;
  int l  = threadIdx.x & 31;
  int gw = blockIdx.x * 4 + w;
  int qt = gw & (SS / 16 - 1);
  int bh = gw >> 7;
  int b  = bh >> 4;
  int h  = bh & 15;
  int qbase = qt * 16;
  const size_t bhS = (size_t)bh * SS;

  v16bf Qa[2];
#pragma unroll
  for (int dc = 0; dc < 2; ++dc)
    Qa[dc] = load_tile(Qh + (bhS + qbase) * DK + dc * 32, DK);

  v8f O[4] = {};
  float mrun[8], lrun[8];
#pragma unroll
  for (int j = 0; j < 8; ++j) { mrun[j] = -3.0e38f; lrun[j] = 0.f; }

  int col  = l & 15;
  int rb16 = l >> 4;

  for (int kb = 0; kb < SS; kb += 32) {
    v8f Sc[2] = {};
#pragma unroll
    for (int nb = 0; nb < 2; ++nb)
#pragma unroll
      for (int dc = 0; dc < 2; ++dc) {
        v16bf Kb = load_tile(Kh + (bhS + kb + nb * 16) * DK + dc * 32, DK);
        Sc[nb] = wmma_bf16(Qa[dc], Kb, Sc[nb]);
      }

    bool m0 = mask[b * SS + kb + col] != 0;
    bool m1 = mask[b * SS + kb + col + 16] != 0;

    float p0[8], p1[8], alpha[8];
#pragma unroll
    for (int j = 0; j < 8; ++j) {
      float s0 = m0 ? -10000.f : Sc[0][j];
      float s1 = m1 ? -10000.f : Sc[1][j];
      float mx = fmaxf(s0, s1);
      mx = fmaxf(mx, __shfl_xor(mx, 1, 32));
      mx = fmaxf(mx, __shfl_xor(mx, 2, 32));
      mx = fmaxf(mx, __shfl_xor(mx, 4, 32));
      mx = fmaxf(mx, __shfl_xor(mx, 8, 32));
      float mnew = fmaxf(mrun[j], mx);
      alpha[j] = __expf(mrun[j] - mnew);
      mrun[j]  = mnew;
      p0[j] = __expf(s0 - mnew);
      p1[j] = __expf(s1 - mnew);
      float ps = p0[j] + p1[j];
      ps += __shfl_xor(ps, 1, 32);
      ps += __shfl_xor(ps, 2, 32);
      ps += __shfl_xor(ps, 4, 32);
      ps += __shfl_xor(ps, 8, 32);
      lrun[j] = lrun[j] * alpha[j] + ps;
    }
#pragma unroll
    for (int t = 0; t < 4; ++t)
#pragma unroll
      for (int j = 0; j < 8; ++j) O[t][j] *= alpha[j];

    // P (16x32 f32, C-layout) -> bf16 -> LDS column-major -> A-layout regs
    {
      uint4 pk0, pk1;
      unsigned int* u0 = (unsigned int*)&pk0;
      unsigned int* u1 = (unsigned int*)&pk1;
#pragma unroll
      for (int jj = 0; jj < 4; ++jj) {
        u0[jj] = (unsigned int)f32_to_bf16(p0[2 * jj]) |
                 ((unsigned int)f32_to_bf16(p0[2 * jj + 1]) << 16);
        u1[jj] = (unsigned int)f32_to_bf16(p1[2 * jj]) |
                 ((unsigned int)f32_to_bf16(p1[2 * jj + 1]) << 16);
      }
      *(uint4*)&Plds[w][col][rb16 * 8]      = pk0;
      *(uint4*)&Plds[w][16 + col][rb16 * 8] = pk1;
    }
    __syncthreads();
    v16bf Pa;
    {
      ABu u;
      int r  = l & 15;
      int c0 = (l >> 4) * 8;
#pragma unroll
      for (int jj = 0; jj < 4; ++jj) {
        u.u[jj] = (unsigned int)Plds[w][c0 + 2 * jj][r] |
                  ((unsigned int)Plds[w][c0 + 2 * jj + 1][r] << 16);
        u.u[4 + jj] = (unsigned int)Plds[w][c0 + 16 + 2 * jj][r] |
                      ((unsigned int)Plds[w][c0 + 17 + 2 * jj][r] << 16);
      }
      Pa = u.v;
    }
    __syncthreads();

#pragma unroll
    for (int t = 0; t < 4; ++t) {
      v16bf Vb = load_tile(VhT + ((size_t)bh * DK + t * 16) * SS + kb, SS);
      O[t] = wmma_bf16(Pa, Vb, O[t]);
    }
  }

  float inv[8];
#pragma unroll
  for (int j = 0; j < 8; ++j) inv[j] = lrun[j] > 0.f ? 1.f / lrun[j] : 0.f;
#pragma unroll
  for (int t = 0; t < 4; ++t)
#pragma unroll
    for (int j = 0; j < 8; ++j) {
      int row = qbase + rb16 * 8 + j;
      int d   = t * 16 + col;
      X[((size_t)(b * SS) + row) * NFEAT + h * DK + d] =
          f32_to_bf16(O[t][j] * inv[j]);
    }
}

extern "C" void kernel_launch(void* const* d_in, const int* in_sizes, int n_in,
                              void* d_out, int out_size, void* d_ws, size_t ws_size,
                              hipStream_t stream) {
  const float* query = (const float*)d_in[0];
  const float* key_  = (const float*)d_in[1];
  const float* value = (const float*)d_in[2];
  const unsigned char* mask = (const unsigned char*)d_in[3];
  const float* w_q = (const float*)d_in[4];
  const float* b_q = (const float*)d_in[5];
  const float* w_k = (const float*)d_in[6];
  const float* b_k = (const float*)d_in[7];
  const float* w_v = (const float*)d_in[8];
  const float* b_v = (const float*)d_in[9];
  const float* w_o = (const float*)d_in[10];
  const float* b_o = (const float*)d_in[11];
  float* out = (float*)d_out;

  const size_t XN = (size_t)BB * SS * NFEAT;   // 4,194,304
  const size_t WN = (size_t)NFEAT * NFEAT;     // 1,048,576

  char* ws = (char*)d_ws;
  unsigned short* qb  = (unsigned short*)ws; ws += XN * 2;
  unsigned short* kb  = (unsigned short*)ws; ws += XN * 2;
  unsigned short* vb  = (unsigned short*)ws; ws += XN * 2;
  unsigned short* wqb = (unsigned short*)ws; ws += WN * 2;
  unsigned short* wkb = (unsigned short*)ws; ws += WN * 2;
  unsigned short* wvb = (unsigned short*)ws; ws += WN * 2;
  unsigned short* wob = (unsigned short*)ws; ws += WN * 2;
  unsigned short* Qh  = (unsigned short*)ws; ws += XN * 2;
  unsigned short* Kh  = (unsigned short*)ws; ws += XN * 2;
  unsigned short* VhT = (unsigned short*)ws; ws += XN * 2;
  unsigned short* Xb  = (unsigned short*)ws; ws += XN * 2;

  cvt_f32_bf16<<<2048, 256, 0, stream>>>(query, qb, (int)XN);
  cvt_f32_bf16<<<2048, 256, 0, stream>>>(key_,  kb, (int)XN);
  cvt_f32_bf16<<<2048, 256, 0, stream>>>(value, vb, (int)XN);
  cvt_f32_bf16<<<1024, 256, 0, stream>>>(w_q, wqb, (int)WN);
  cvt_f32_bf16<<<1024, 256, 0, stream>>>(w_k, wkb, (int)WN);
  cvt_f32_bf16<<<1024, 256, 0, stream>>>(w_v, wvb, (int)WN);
  cvt_f32_bf16<<<1024, 256, 0, stream>>>(w_o, wob, (int)WN);

  const int proj_blocks = (BB * SS / 64) * (NFEAT / 64);   // 1024
  proj_wmma<<<proj_blocks, 128, 0, stream>>>(qb, wqb, b_q, Qh,  nullptr, 0);
  proj_wmma<<<proj_blocks, 128, 0, stream>>>(kb, wkb, b_k, Kh,  nullptr, 1);
  proj_wmma<<<proj_blocks, 128, 0, stream>>>(vb, wvb, b_v, VhT, nullptr, 2);

  attn_fwd<<<(BB * NHEAD * SS / 16) / 4, 128, 0, stream>>>(Qh, Kh, VhT, mask, Xb);

  proj_wmma<<<proj_blocks, 128, 0, stream>>>(Xb, wob, b_o, nullptr, out, 3);
}
